// Encoder_80814104642078
// MI455X (gfx1250) — matrix operand order repaired and checked by reading, not compiled, not run on hardware
//
#include <hip/hip_runtime.h>
#include <hip/hip_bf16.h>
#include <stdint.h>

// ---------------------------------------------------------------------------
// CDNA5 (gfx1250) GAE encoder:
//   H1 = feat @ W1            (bf16 WMMA, f32 accum)
//   H1a = SpMM(adj, H1)       (float4 gather + global_atomic_add_f32 scatter)
//   H2 = H1a @ W2             (bf16 WMMA)
//   z  = SpMM(adj, H2)        -> d_out[0 : N*64]
//   A_rec = sigmoid(z @ z^T)  (bf16 WMMA, store-bandwidth bound)
//                             -> d_out[N*64 : N*64 + N*N]
// ---------------------------------------------------------------------------

typedef __attribute__((ext_vector_type(16))) __bf16 bf16x16;
typedef __attribute__((ext_vector_type(8)))  float  v8f;
typedef unsigned short u16;

// ---- helpers --------------------------------------------------------------

__device__ __forceinline__ u16 f32_to_bf16_rne(float f) {
    unsigned int u = __float_as_uint(f);
    u += 0x7FFFu + ((u >> 16) & 1u);           // round-to-nearest-even
    return (u16)(u >> 16);
}

// Build a 16-element bf16 fragment from two 16-byte chunks (global_load_b128 x2)
__device__ __forceinline__ bf16x16 load_frag(const u16* p0, const u16* p1) {
    bf16x16 r;
    reinterpret_cast<uint4*>(&r)[0] = *reinterpret_cast<const uint4*>(p0);
    reinterpret_cast<uint4*>(&r)[1] = *reinterpret_cast<const uint4*>(p1);
    return r;
}

// ---- elementwise kernels --------------------------------------------------

__global__ void k_f32_to_bf16(const float* __restrict__ in,
                              u16* __restrict__ out, long long n) {
    long long i = (long long)blockIdx.x * blockDim.x + threadIdx.x;
    if (i < n) out[i] = f32_to_bf16_rne(in[i]);
}

// W [R,C] f32 row-major  ->  WT [C,R] bf16 row-major (row n of WT = column n of W)
__global__ void k_transpose_bf16(const float* __restrict__ W,
                                 u16* __restrict__ WT, int R, int C) {
    long long i = (long long)blockIdx.x * blockDim.x + threadIdx.x;
    if (i >= (long long)R * C) return;
    int r = (int)(i / C), c = (int)(i % C);
    WT[(size_t)c * R + r] = f32_to_bf16_rne(W[i]);
}

__global__ void k_zero_f32(float* __restrict__ p, long long n) {
    long long i = (long long)blockIdx.x * blockDim.x + threadIdx.x;
    if (i < n) p[i] = 0.0f;
}

// ---- SpMM: out[rows[e], :] += vals[e] * x[cols[e], :]  (D multiple of 4) ---

__global__ void k_spmm_scatter(const int* __restrict__ rows,
                               const int* __restrict__ cols,
                               const float* __restrict__ vals,
                               const float* __restrict__ x,
                               float* __restrict__ out,
                               int E, int D) {
    const int per_edge = D >> 2;                       // float4 chunks per edge
    long long gid = (long long)blockIdx.x * blockDim.x + threadIdx.x;
    if (gid >= (long long)E * per_edge) return;
    int e  = (int)(gid / per_edge);
    int c4 = (int)(gid % per_edge) << 2;
    int r = rows[e], c = cols[e];
    float v = vals[e];
    float4 xv = *reinterpret_cast<const float4*>(x + (size_t)c * D + c4);
    float* o = out + (size_t)r * D + c4;
    atomicAdd(o + 0, v * xv.x);
    atomicAdd(o + 1, v * xv.y);
    atomicAdd(o + 2, v * xv.z);
    atomicAdd(o + 3, v * xv.w);
}

// ---- bf16 WMMA GEMM: C[M,N] = A[M,K] * BT[N,K]^T  (all dims %16/%32 == 0) --
// One wave computes one 16x16 C tile; 4 waves / 128-thread block.

__global__ void __launch_bounds__(128)
k_gemm_bf16_wmma(const u16* __restrict__ A, const u16* __restrict__ BT,
                 float* __restrict__ C, int M, int N, int K) {
    const int wave    = blockIdx.x * (blockDim.x >> 5) + (threadIdx.x >> 5);
    const int lane    = threadIdx.x & 31;
    const int tiles_n = N >> 4;
    const int tm = wave / tiles_n;
    const int tn = wave % tiles_n;
    if (tm >= (M >> 4)) return;                        // wave-uniform exit

    const int l    = lane & 15;                        // A row / B column
    const int half = lane >> 4;

    const u16* arow = A  + (size_t)(tm * 16 + l) * K;
    const u16* brow = BT + (size_t)(tn * 16 + l) * K;

    v8f acc = {};
    for (int k0 = 0; k0 < K; k0 += 32) {
        __builtin_prefetch(arow + k0 + 64, 0, 0);
        __builtin_prefetch(brow + k0 + 64, 0, 0);
        // A: lane holds row l, K = k0 + half*8 + {0..7}  and  +16
        bf16x16 a = load_frag(arow + k0 + half * 8,
                              arow + k0 + 16 + half * 8);
        // B: lane holds column l, K = k0 + half*8 + {0..7}  and  +16
        bf16x16 b = load_frag(brow + k0 + half * 8,
                              brow + k0 + 16 + half * 8);
        acc = __builtin_amdgcn_wmma_f32_16x16x32_bf16(
                  false, a, false, b, (short)0, acc, false, false);
    }

    // D layout: VGPR i -> M = i + 8*half, N = l
    float* crow = C + (size_t)(tm * 16 + half * 8) * N + (tn * 16 + l);
    #pragma unroll
    for (int i = 0; i < 8; ++i) crow[(size_t)i * N] = acc[i];
}

// ---- decoder: A_rec = sigmoid(Z Z^T), Z[Nn,64] bf16 ------------------------

__global__ void __launch_bounds__(128)
k_decoder_wmma(const u16* __restrict__ Zbf, float* __restrict__ Arec, int Nn) {
    const int wave  = blockIdx.x * (blockDim.x >> 5) + (threadIdx.x >> 5);
    const int lane  = threadIdx.x & 31;
    const int tiles = Nn >> 4;                         // 1250
    const int tm = wave / tiles;
    const int tn = wave % tiles;
    if (tm >= tiles) return;

    const int l    = lane & 15;
    const int half = lane >> 4;
    const int K    = 64;

    const u16* arow = Zbf + (size_t)(tm * 16 + l) * K; // row l of A-tile
    const u16* brow = Zbf + (size_t)(tn * 16 + l) * K; // col l of B = row of Z

    v8f acc = {};
    #pragma unroll
    for (int k0 = 0; k0 < 64; k0 += 32) {
        bf16x16 a = load_frag(arow + k0 + half * 8,
                              arow + k0 + 16 + half * 8);
        bf16x16 b = load_frag(brow + k0 + half * 8,
                              brow + k0 + 16 + half * 8);
        acc = __builtin_amdgcn_wmma_f32_16x16x32_bf16(
                  false, a, false, b, (short)0, acc, false, false);
    }

    float* crow = Arec + (size_t)(tm * 16 + half * 8) * Nn + (tn * 16 + l);
    #pragma unroll
    for (int i = 0; i < 8; ++i) {
        float s = 1.0f / (1.0f + __expf(-acc[i]));     // sigmoid
        crow[(size_t)i * Nn] = s;
    }
}

// ---------------------------------------------------------------------------

extern "C" void kernel_launch(void* const* d_in, const int* in_sizes, int n_in,
                              void* d_out, int out_size, void* d_ws, size_t ws_size,
                              hipStream_t stream) {
    const float* feat = (const float*)d_in[0];
    const int*   rows = (const int*)  d_in[1];
    const int*   cols = (const int*)  d_in[2];
    const float* vals = (const float*)d_in[3];
    const float* W1   = (const float*)d_in[4];
    const float* W2   = (const float*)d_in[5];

    const int F0 = 512, H = 256, O = 64;
    const int Nn = in_sizes[0] / F0;                   // 20000
    const int E  = in_sizes[1];                        // 640000

    // ---- carve workspace (256B aligned) ----
    char* wp = (char*)d_ws;
    auto carve = [&](size_t bytes) {
        char* p = wp;
        wp += (bytes + 255) & ~(size_t)255;
        return p;
    };
    u16*   featbf = (u16*)  carve((size_t)Nn * F0 * sizeof(u16));
    u16*   W1T    = (u16*)  carve((size_t)H  * F0 * sizeof(u16));
    u16*   W2T    = (u16*)  carve((size_t)O  * H  * sizeof(u16));
    float* H1     = (float*)carve((size_t)Nn * H  * sizeof(float));
    float* H1a    = (float*)carve((size_t)Nn * H  * sizeof(float));
    u16*   H1abf  = (u16*)  carve((size_t)Nn * H  * sizeof(u16));
    float* H2     = (float*)carve((size_t)Nn * O  * sizeof(float));
    u16*   zbf    = (u16*)  carve((size_t)Nn * O  * sizeof(u16));

    float* z    = (float*)d_out;                       // first N*64 outputs
    float* Arec = (float*)d_out + (size_t)Nn * O;      // then N*N outputs

    const int T = 256;
    auto gridFor = [&](long long n) { return (unsigned)((n + T - 1) / T); };

    // 1) convert feat and weights to bf16 (weights pre-transposed to [N,K])
    k_f32_to_bf16   <<<gridFor((long long)Nn * F0), T, 0, stream>>>(feat, featbf, (long long)Nn * F0);
    k_transpose_bf16<<<gridFor((long long)F0 * H),  T, 0, stream>>>(W1, W1T, F0, H);
    k_transpose_bf16<<<gridFor((long long)H  * O),  T, 0, stream>>>(W2, W2T, H, O);

    // 2) H1 = feat @ W1
    {
        unsigned waves = (unsigned)((Nn >> 4) * (H >> 4));
        k_gemm_bf16_wmma<<<(waves + 3) / 4, 128, 0, stream>>>(featbf, W1T, H1, Nn, H, F0);
    }

    // 3) H1a = SpMM(adj, H1)
    k_zero_f32<<<gridFor((long long)Nn * H), T, 0, stream>>>(H1a, (long long)Nn * H);
    k_spmm_scatter<<<gridFor((long long)E * (H >> 2)), T, 0, stream>>>(rows, cols, vals, H1, H1a, E, H);

    // 4) H2 = H1a @ W2
    k_f32_to_bf16<<<gridFor((long long)Nn * H), T, 0, stream>>>(H1a, H1abf, (long long)Nn * H);
    {
        unsigned waves = (unsigned)((Nn >> 4) * (O >> 4));
        k_gemm_bf16_wmma<<<(waves + 3) / 4, 128, 0, stream>>>(H1abf, W2T, H2, Nn, O, H);
    }

    // 5) z = SpMM(adj, H2)  -> d_out
    k_zero_f32<<<gridFor((long long)Nn * O), T, 0, stream>>>(z, (long long)Nn * O);
    k_spmm_scatter<<<gridFor((long long)E * (O >> 2)), T, 0, stream>>>(rows, cols, vals, H2, z, E, O);

    // 6) A_rec = sigmoid(z z^T)
    k_f32_to_bf16<<<gridFor((long long)Nn * O), T, 0, stream>>>(z, zbf, (long long)Nn * O);
    {
        long long waves = (long long)(Nn >> 4) * (Nn >> 4);   // 1,562,500
        k_decoder_wmma<<<(unsigned)((waves + 3) / 4), 128, 0, stream>>>(zbf, Arec, Nn);
    }
}
